// GCNModel_7859790152118
// MI455X (gfx1250) — matrix operand order repaired and checked by reading, not compiled, not run on hardware
//
#include <hip/hip_runtime.h>
#include <hip/hip_bf16.h>

// GCN forward, 3 layers, D=128, on MI455X (gfx1250, wave32).
//   per layer: h = relu_opt(x) @ W[l]        (bf16 WMMA, fp32 accumulate)
//              xn = b[l] + dinv^2 * h        (fused into GEMM epilogue)
//              xn += scatter_add(dinv[r]*dinv[c]*h[r] -> c)   (wave-per-edge atomics)
//   final: relu(d_out)
// Feature matrices (51.2 MB) are L2-resident (192 MB L2), so gather/scatter runs
// out of L2; GEMM hits v_wmma_f32_16x16x32_bf16 with LDS-staged, pre-swizzled W.

#define D        128
#define NLAYERS  3

typedef __attribute__((ext_vector_type(16))) __bf16 v16bf;
typedef __attribute__((ext_vector_type(8)))  float  v8f;

// ---------------------------------------------------------------- degrees ---
__global__ __launch_bounds__(256) void deg_init_kernel(float* __restrict__ deg, int n) {
    int i = blockIdx.x * 256 + threadIdx.x;
    if (i < n) deg[i] = 1.0f;               // self-loop contributes 1
}

__global__ __launch_bounds__(256) void deg_count_kernel(const int* __restrict__ ei,
                                                        float* __restrict__ deg, int nE) {
    int e = blockIdx.x * 256 + threadIdx.x;
    if (e < nE) {
        int c = ei[nE + e];                 // edge_index[1][e] (target)
        atomicAdd(&deg[c], 1.0f);
    }
}

__global__ __launch_bounds__(256) void deg_finalize_kernel(float* __restrict__ deg, int n) {
    int i = blockIdx.x * 256 + threadIdx.x;
    if (i < n) deg[i] = rsqrtf(deg[i]);     // deg >= 1 always (self-loop)
}

// ------------------------------------------------------------------- GEMM ---
// h[N x 128] = relu_opt(x)[N x 128] @ W[128 x 128]   (bf16 in, fp32 accum)
// Epilogue also emits xn = bias + dinv^2 * h (scatter-target init, self-loop
// message and bias fused). Block = 256 threads = 8 waves; each wave owns a
// 16-row x 128-col strip (8 accumulator tiles, A fragment reused 8x per K step).
__global__ __launch_bounds__(256) void gemm_bf16_wmma_kernel(const float* __restrict__ x,
                                                             const float* __restrict__ W,
                                                             const float* __restrict__ bias,
                                                             const float* __restrict__ dinv,
                                                             float* __restrict__ h,
                                                             float* __restrict__ xn,
                                                             int nNodes, int doRelu) {
    // W staged in LDS, pre-swizzled into B-fragment order:
    // idx = ((kt*8 + nt)*32 + lane)*16 + e  ->  W[K][col],
//   K = kt*32 + 8*(lane>=16) + (e&7) + 16*(e>=8),  col = nt*16 + (lane&15)
    __shared__ __attribute__((aligned(32))) __bf16 Wl[4 * 8 * 32 * 16]; // 32 KB

    int tid = threadIdx.x;
    for (int j = tid; j < D * D; j += 256) {          // coalesced fp32 reads
        int K    = j >> 7;
        int col  = j & 127;
        int kt   = K >> 5;
        int koff = (K >> 3) & 1;
        int e    = (K & 7) | (((K >> 4) & 1) << 3);
        int nt   = col >> 4;
        int lane = (col & 15) + (koff << 4);
        Wl[(((kt << 3) + nt) * 32 + lane) * 16 + e] = (__bf16)W[j];
    }
    __syncthreads();

    int wave    = tid >> 5;
    int lane    = tid & 31;
    int half    = lane >> 4;                 // 0: lanes 0-15, 1: lanes 16-31
    int rowBase = blockIdx.x * 128 + wave * 16;
    int r       = rowBase + (lane & 15);     // A-matrix row for this lane
    bool valid  = (r < nNodes);
    const float* xr = x + (size_t)r * D;
    int khalf   = half << 3;                 // 0 or 8
    int colLane = lane & 15;

    // per-lane bias values for the 8 column tiles
    float bv[8];
    #pragma unroll
    for (int nt = 0; nt < 8; ++nt) bv[nt] = bias[nt * 16 + colLane];

    v8f acc[8];
    #pragma unroll
    for (int nt = 0; nt < 8; ++nt) {
        #pragma unroll
        for (int v = 0; v < 8; ++v) acc[nt][v] = 0.0f;
    }

    #pragma unroll
    for (int kt = 0; kt < 4; ++kt) {         // K = 128 in 4 steps of 32
        int k0 = kt << 5;
        v16bf a;
        #pragma unroll
        for (int q = 0; q < 16; ++q) a[q] = (__bf16)0.0f;
        if (valid) {
            // ISA 16-bit A 16x32 layout: e<8 -> K=khalf+e ; e>=8 -> K=16+khalf+(e-8)
            float4 p0 = *(const float4*)(xr + k0 + khalf);
            float4 p1 = *(const float4*)(xr + k0 + khalf + 4);
            float4 p2 = *(const float4*)(xr + k0 + 16 + khalf);
            float4 p3 = *(const float4*)(xr + k0 + 16 + khalf + 4);
            if (doRelu) {                    // previous layer's activation, fused
                p0.x = fmaxf(p0.x, 0.f); p0.y = fmaxf(p0.y, 0.f);
                p0.z = fmaxf(p0.z, 0.f); p0.w = fmaxf(p0.w, 0.f);
                p1.x = fmaxf(p1.x, 0.f); p1.y = fmaxf(p1.y, 0.f);
                p1.z = fmaxf(p1.z, 0.f); p1.w = fmaxf(p1.w, 0.f);
                p2.x = fmaxf(p2.x, 0.f); p2.y = fmaxf(p2.y, 0.f);
                p2.z = fmaxf(p2.z, 0.f); p2.w = fmaxf(p2.w, 0.f);
                p3.x = fmaxf(p3.x, 0.f); p3.y = fmaxf(p3.y, 0.f);
                p3.z = fmaxf(p3.z, 0.f); p3.w = fmaxf(p3.w, 0.f);
            }
            a[0]  = (__bf16)p0.x; a[1]  = (__bf16)p0.y; a[2]  = (__bf16)p0.z; a[3]  = (__bf16)p0.w;
            a[4]  = (__bf16)p1.x; a[5]  = (__bf16)p1.y; a[6]  = (__bf16)p1.z; a[7]  = (__bf16)p1.w;
            a[8]  = (__bf16)p2.x; a[9]  = (__bf16)p2.y; a[10] = (__bf16)p2.z; a[11] = (__bf16)p2.w;
            a[12] = (__bf16)p3.x; a[13] = (__bf16)p3.y; a[14] = (__bf16)p3.z; a[15] = (__bf16)p3.w;
        }
        #pragma unroll
        for (int nt = 0; nt < 8; ++nt) {     // A fragment reused 8x
            v16bf b = *(const v16bf*)(Wl + (((kt << 3) + nt) * 32 + lane) * 16);
            acc[nt] = __builtin_amdgcn_wmma_f32_16x16x32_bf16(
                false, a, false, b, (short)0, acc[nt], false, false);
        }
    }

    // C/D layout: VGPR v -> row (v + 8*half), col = lane&15 within the tile.
    // Store h and the fused scatter-target init xn = bias + dinv^2 * h.
    #pragma unroll
    for (int v = 0; v < 8; ++v) {
        int rr = rowBase + (half << 3) + v;
        if (rr < nNodes) {
            float s = dinv[rr];
            s = s * s;
            float* hp = h  + (size_t)rr * D + colLane;
            float* xp = xn + (size_t)rr * D + colLane;
            #pragma unroll
            for (int nt = 0; nt < 8; ++nt) {
                float hvv = acc[nt][v];
                hp[nt * 16] = hvv;
                xp[nt * 16] = fmaf(hvv, s, bv[nt]);
            }
        }
    }
}

// ------------------------------------------------------------ scatter-add ---
// One wave32 per edge: lane L handles features [4L, 4L+4). Gather is one
// coalesced 512B read of h[row]; scatter is 4 global_atomic_add_f32 per lane.
// Both h and xn are L2-resident on MI455X.
__global__ __launch_bounds__(256) void scatter_kernel(const int* __restrict__ ei,
                                                      const float* __restrict__ h,
                                                      const float* __restrict__ dinv,
                                                      float* __restrict__ xn, int nE) {
    int e = blockIdx.x * 8 + (threadIdx.x >> 5);
    if (e >= nE) return;
    int lane = threadIdx.x & 31;
    int r = ei[e];            // source
    int c = ei[nE + e];       // target
    float nrm = dinv[r] * dinv[c];   // same-address loads broadcast in hardware
    float4 v = *(const float4*)(h + (size_t)r * D + lane * 4);
    float* dst = xn + (size_t)c * D + lane * 4;
    atomicAdd(dst + 0, v.x * nrm);
    atomicAdd(dst + 1, v.y * nrm);
    atomicAdd(dst + 2, v.z * nrm);
    atomicAdd(dst + 3, v.w * nrm);
}

// ------------------------------------------------------ final-layer ReLU ----
__global__ __launch_bounds__(256) void relu_kernel(float* __restrict__ x, int n) {
    int t = blockIdx.x * 256 + threadIdx.x;
    if (t >= n * (D / 4)) return;
    float4 v = ((float4*)x)[t];
    v.x = fmaxf(v.x, 0.0f);
    v.y = fmaxf(v.y, 0.0f);
    v.z = fmaxf(v.z, 0.0f);
    v.w = fmaxf(v.w, 0.0f);
    ((float4*)x)[t] = v;
}

// ----------------------------------------------------------------- launch ---
extern "C" void kernel_launch(void* const* d_in, const int* in_sizes, int n_in,
                              void* d_out, int out_size, void* d_ws, size_t ws_size,
                              hipStream_t stream) {
    const int*   ei  = (const int*)d_in[0];     // edge_index [2, E]
    const float* emb = (const float*)d_in[1];   // [N, 128]
    const float* Ws  = (const float*)d_in[2];   // [3, 128, 128]
    const float* bs  = (const float*)d_in[3];   // [3, 128]
    float* out = (float*)d_out;

    const int nE = in_sizes[0] / 2;
    const int nN = in_sizes[1] / D;

    // workspace: dinv | h | x   (all fp32)
    float* wsf  = (float*)d_ws;
    int nPad    = ((nN + 1023) / 1024) * 1024;
    float* dinv = wsf;
    float* hbuf = wsf + nPad;
    float* xbuf = hbuf + (size_t)nN * D;

    int nodeBlocks = (nN + 255) / 256;
    int edgeBlocks = (nE + 255) / 256;
    int featBlocks = (nN * (D / 4) + 255) / 256;
    int gemmBlocks = (nN + 127) / 128;
    int scatBlocks = (nE + 7) / 8;

    // symmetric normalization: dinv = 1/sqrt(deg + 1)
    deg_init_kernel    <<<nodeBlocks, 256, 0, stream>>>(dinv, nN);
    deg_count_kernel   <<<edgeBlocks, 256, 0, stream>>>(ei, dinv, nE);
    deg_finalize_kernel<<<nodeBlocks, 256, 0, stream>>>(dinv, nN);

    const float* xcur = emb;
    for (int l = 0; l < NLAYERS; ++l) {
        float* xnext = (l == NLAYERS - 1) ? out : xbuf;
        // In-place xcur==xnext (layer 2) is safe: each block reads only its own
        // 128 rows for A fragments, all reads precede epilogue writes.
        gemm_bf16_wmma_kernel<<<gemmBlocks, 256, 0, stream>>>(
            xcur, Ws + l * D * D, bs + l * D, dinv, hbuf, xnext, nN, l > 0);
        scatter_kernel<<<scatBlocks, 256, 0, stream>>>(ei, hbuf, dinv, xnext, nE);
        xcur = xnext;
    }
    relu_kernel<<<featBlocks, 256, 0, stream>>>(out, nN);
}
